// MultiBoxLoss_29566554865720
// MI455X (gfx1250) — compile-verified, matrix operand order checked by reading, not run on hardware
//
#include <hip/hip_runtime.h>
#include <hip/hip_bf16.h>
#include <math.h>

#define NIMG   32
#define NPRI   8732
#define NOBJ   32
#define NCLS   81
#define THRESH 0.5f
#define NEGPOS 3
#define CE_ROWS 96
#define SORT_N 16384

typedef int          v8i  __attribute__((ext_vector_type(8)));
typedef unsigned int u32x4 __attribute__((ext_vector_type(4)));
typedef int          i32x4 __attribute__((ext_vector_type(4)));
typedef int          i32x8 __attribute__((ext_vector_type(8)));

// ---------------------------------------------------------------- init
__global__ void k_init(float* acc, int* npos, unsigned long long* objbest) {
  int t = threadIdx.x;
  if (t < 3) acc[t] = 0.f;
  if (t <= NIMG) npos[t] = 0;
  for (int idx = t; idx < NIMG * NOBJ; idx += blockDim.x) objbest[idx] = 0ull;
}

// ---------------------------------------------------------------- match: IoU, per-prior argmax, per-object best prior
__global__ void k_match(const float* __restrict__ boxes, const float* __restrict__ priors,
                        int* __restrict__ prior_obj, float* __restrict__ prior_iou,
                        unsigned long long* __restrict__ objbest) {
  __shared__ unsigned long long sbest[NOBJ];
  __shared__ float sbox[NOBJ * 4];
  int tid = threadIdx.x;
  int i = blockIdx.y;
  int p = blockIdx.x * blockDim.x + tid;
  if (tid < NOBJ) sbest[tid] = 0ull;
  for (int idx = tid; idx < NOBJ * 4; idx += blockDim.x) sbox[idx] = boxes[i * NOBJ * 4 + idx];
  __syncthreads();
  if (p < NPRI) {
    float pcx = priors[p * 4 + 0], pcy = priors[p * 4 + 1];
    float pw  = priors[p * 4 + 2], ph  = priors[p * 4 + 3];
    float px0 = pcx - pw * 0.5f, py0 = pcy - ph * 0.5f;
    float px1 = pcx + pw * 0.5f, py1 = pcy + ph * 0.5f;
    float parea = (px1 - px0) * (py1 - py0);
    float best = -1.f; int besto = 0;
    for (int o = 0; o < NOBJ; ++o) {
      float bx0 = sbox[o * 4 + 0], by0 = sbox[o * 4 + 1];
      float bx1 = sbox[o * 4 + 2], by1 = sbox[o * 4 + 3];
      float w = fmaxf(fminf(px1, bx1) - fmaxf(px0, bx0), 0.f);
      float h = fmaxf(fminf(py1, by1) - fmaxf(py0, by0), 0.f);
      float inter = w * h;
      float barea = (bx1 - bx0) * (by1 - by0);
      float iou = inter / (parea + barea - inter);
      if (iou > best) { best = iou; besto = o; }          // first max wins (jnp.argmax)
      unsigned long long key = ((unsigned long long)__float_as_uint(iou) << 32)
                             | (unsigned long long)(0xFFFFFFFFu - (unsigned)p); // ties -> smallest p
      atomicMax(&sbest[o], key);
    }
    prior_obj[i * NPRI + p] = besto;
    prior_iou[i * NPRI + p] = best;
  }
  __syncthreads();
  if (tid < NOBJ) atomicMax(&objbest[i * NOBJ + tid], sbest[tid]);
}

// ---------------------------------------------------------------- force each object to own its best prior
__global__ void k_override(const unsigned long long* __restrict__ objbest,
                           int* __restrict__ prior_obj, float* __restrict__ prior_iou) {
  int i = threadIdx.x;
  if (i < NIMG) {
    for (int o = 0; o < NOBJ; ++o) {
      unsigned long long key = objbest[i * NOBJ + o];
      unsigned p = 0xFFFFFFFFu - (unsigned)(key & 0xFFFFFFFFull);
      prior_obj[i * NPRI + p] = o;
      prior_iou[i * NPRI + p] = 1.0f;
    }
  }
}

// ---------------------------------------------------------------- loc loss + class labels + WMMA positive count
__global__ void k_loc(const float* __restrict__ pred_off, const float* __restrict__ boxes,
                      const int* __restrict__ labels, const float* __restrict__ priors,
                      const int* __restrict__ prior_obj, const float* __restrict__ prior_iou,
                      int* __restrict__ prior_cls, float* __restrict__ acc,
                      int* __restrict__ npos) {
  __shared__ unsigned char flags[256];
  __shared__ float s_loc;
  __shared__ int s_cnt;
  int tid = threadIdx.x;
  if (tid == 0) { s_loc = 0.f; s_cnt = 0; }
  int i = blockIdx.y;
  int p = blockIdx.x * 256 + tid;
  int pos = 0;
  float loc = 0.f;
  if (p < NPRI) {
    int   o   = prior_obj[i * NPRI + p];
    float iou = prior_iou[i * NPRI + p];
    int   cls = (iou < THRESH) ? 0 : labels[i * NOBJ + o];
    prior_cls[i * NPRI + p] = cls;
    pos = (cls != 0);
    if (pos) {
      const float* b = &boxes[(i * NOBJ + o) * 4];
      float bcx = (b[0] + b[2]) * 0.5f, bcy = (b[1] + b[3]) * 0.5f;
      float bw = b[2] - b[0], bh = b[3] - b[1];
      float pcx = priors[p * 4 + 0], pcy = priors[p * 4 + 1];
      float pw  = priors[p * 4 + 2], ph  = priors[p * 4 + 3];
      float g0 = (bcx - pcx) / (pw * 0.1f);
      float g1 = (bcy - pcy) / (ph * 0.1f);
      float g2 = logf(bw / pw) * 5.f;
      float g3 = logf(bh / ph) * 5.f;
      const float* po = &pred_off[((size_t)i * NPRI + p) * 4];
      loc = fabsf(po[0] - g0) + fabsf(po[1] - g1) + fabsf(po[2] - g2) + fabsf(po[3] - g3);
    }
  }
  flags[tid] = (unsigned char)pos;
  __syncthreads();
  if (loc != 0.f) atomicAdd(&s_loc, loc);
  // Exact block positive-count on the matrix unit: A = 256 byte-flags, B = ones.
  // D[m][n] = rowsum(A_m) replicated across 16 columns -> sum(all D)/16 = sum(flags).
  if (tid < 32) {                                    // wave 0 fully active -> EXEC all 1s
    const int* fp = (const int*)flags;               // 64 dwords of flag bytes
    v8i a = { fp[tid * 2], fp[tid * 2 + 1], 0, 0, 0, 0, 0, 0 };
    v8i b = { 0x01010101, 0x01010101, 0x01010101, 0x01010101,
              0x01010101, 0x01010101, 0x01010101, 0x01010101 };
    v8i c = { 0, 0, 0, 0, 0, 0, 0, 0 };
    c = __builtin_amdgcn_wmma_i32_16x16x64_iu8(false, a, false, b, c, false, false);
    int s = c[0] + c[1] + c[2] + c[3] + c[4] + c[5] + c[6] + c[7];
    for (int off = 16; off > 0; off >>= 1) s += __shfl_xor(s, off, 32);
    if (tid == 0) s_cnt = s >> 4;
  }
  __syncthreads();
  if (tid == 0) {
    if (s_loc != 0.f) atomicAdd(&acc[0], s_loc);
    if (s_cnt) { atomicAdd(&npos[i], s_cnt); atomicAdd(&npos[NIMG], s_cnt); }
  }
}

// ---------------------------------------------------------------- fused softmax-CE over the 90MB score stream
__global__ void k_ce(const float* __restrict__ scores, const int* __restrict__ prior_cls,
                     float* __restrict__ neg_ce, float* __restrict__ acc) {
  __shared__ float tile[CE_ROWS * NCLS];
  __shared__ float s_pos;
  int tid = threadIdx.x;
  if (tid == 0) s_pos = 0.f;
  const int total = NIMG * NPRI;
  int row0 = blockIdx.x * CE_ROWS;

#if __has_builtin(__builtin_amdgcn_tensor_load_to_lds)
  if (tid < 32) {   // one wave issues the TDM descriptor (EXEC ignored by TDM)
    unsigned lds_base = (unsigned)(size_t)&tile[0];
    unsigned long long ga = (unsigned long long)(size_t)(scores + (size_t)row0 * NCLS);
    unsigned rows_rem = (unsigned)(total - row0);    // OOB rows read as zero
    u32x4 g0 = { 1u,                                  // count=1, user descriptor
                 lds_base,                            // lds_addr
                 (unsigned)(ga & 0xFFFFFFFFull),      // global_addr[31:0]
                 (unsigned)((ga >> 32) & 0x1FFFFFFull) | (2u << 30) }; // addr[56:32] | type=2
    i32x8 g1 = { (int)(2u << 16),                     // data_size = 4B
                 (int)(((unsigned)NCLS & 0xFFFFu) << 16),            // tensor_dim0 lo16
                 (int)((rows_rem & 0xFFFFu) << 16),                  // dim0 hi16=0 | tensor_dim1 lo16
                 (int)(((rows_rem >> 16) & 0xFFFFu) | ((unsigned)NCLS << 16)), // dim1 hi16 | tile_dim0
                 (int)CE_ROWS,                        // tile_dim1 | tile_dim2=0
                 (int)NCLS,                           // tensor_dim0_stride lo32
                 0, 0 };
    i32x4 z4 = { 0, 0, 0, 0 };
#if defined(__clang_major__) && __clang_major__ >= 23
    i32x8 z8 = { 0, 0, 0, 0, 0, 0, 0, 0 };
    __builtin_amdgcn_tensor_load_to_lds(g0, g1, z4, z4, z8, 0);
#else
    __builtin_amdgcn_tensor_load_to_lds(g0, g1, z4, z4, 0);
#endif
    __builtin_amdgcn_s_wait_tensorcnt(0);
  }
#else
  for (int idx = tid; idx < CE_ROWS * NCLS; idx += blockDim.x) {
    int r = row0 + idx / NCLS;
    tile[idx] = (r < total) ? scores[(size_t)row0 * NCLS + idx] : 0.f;
  }
#endif
  __syncthreads();

  int wave = tid >> 5, lane = tid & 31;
  float posAcc = 0.f;
  for (int rr = wave; rr < CE_ROWS; rr += 8) {
    int r = row0 + rr;
    if (r >= total) break;
    const float* row = &tile[rr * NCLS];
    float v0 = row[lane];
    float v1 = row[lane + 32];
    float v2 = (lane + 64 < NCLS) ? row[lane + 64] : -3.4e38f;
    float m = fmaxf(v0, fmaxf(v1, v2));
    for (int off = 16; off > 0; off >>= 1) m = fmaxf(m, __shfl_xor(m, off, 32));
    float e = expf(v0 - m) + expf(v1 - m) + ((lane + 64 < NCLS) ? expf(v2 - m) : 0.f);
    for (int off = 16; off > 0; off >>= 1) e += __shfl_xor(e, off, 32);
    if (lane == 0) {
      int c = prior_cls[r];
      float ce = (m + logf(e)) - row[c];
      if (c != 0) posAcc += ce;
      neg_ce[r] = (c != 0) ? 0.f : ce;
    }
  }
  if (lane == 0 && posAcc != 0.f) atomicAdd(&s_pos, posAcc);
  __syncthreads();
  if (tid == 0 && s_pos != 0.f) atomicAdd(&acc[1], s_pos);
}

// ---------------------------------------------------------------- hard-negative mining: LDS bitonic sort + top-k sum
__global__ void k_topk(const float* __restrict__ neg_ce, const int* __restrict__ npos,
                       float* __restrict__ acc) {
  __shared__ float s[SORT_N];
  __shared__ float red[32];
  int i = blockIdx.x, tid = threadIdx.x;
  for (int idx = tid; idx < SORT_N; idx += blockDim.x)
    s[idx] = (idx < NPRI) ? neg_ce[(size_t)i * NPRI + idx] : -1.0f;  // pad < 0 <= all CE
  __syncthreads();
  for (int size = 2; size <= SORT_N; size <<= 1) {
    for (int stride = size >> 1; stride > 0; stride >>= 1) {
      for (int t = tid; t < SORT_N; t += blockDim.x) {
        int j = t ^ stride;
        if (j > t) {
          bool up = ((t & size) == 0);
          float a = s[t], b = s[j];
          if ((a > b) == up) { s[t] = b; s[j] = a; }
        }
      }
      __syncthreads();
    }
  }
  int k = NEGPOS * npos[i];
  if (k > NPRI) k = NPRI;
  float sum = 0.f;
  for (int idx = tid; idx < k; idx += blockDim.x) sum += s[SORT_N - 1 - idx]; // largest at end
  for (int off = 16; off > 0; off >>= 1) sum += __shfl_xor(sum, off, 32);
  int lane = tid & 31, w = tid >> 5;
  if (lane == 0) red[w] = sum;
  __syncthreads();
  if (tid == 0) {
    float tot = 0.f;
    for (int x = 0; x < (int)(blockDim.x >> 5); ++x) tot += red[x];
    if (tot != 0.f) atomicAdd(&acc[2], tot);
  }
}

// ---------------------------------------------------------------- combine
__global__ void k_final(const float* __restrict__ acc, const int* __restrict__ npos,
                        float* __restrict__ out) {
  float npt = (float)npos[NIMG];
  out[0] = acc[0] / (npt * 4.f) + (acc[1] + acc[2]) / npt;
}

// ---------------------------------------------------------------- host
extern "C" void kernel_launch(void* const* d_in, const int* in_sizes, int n_in,
                              void* d_out, int out_size, void* d_ws, size_t ws_size,
                              hipStream_t stream) {
  (void)in_sizes; (void)n_in; (void)out_size; (void)ws_size;
  const float* pred_off = (const float*)d_in[0];
  const float* scores   = (const float*)d_in[1];
  const float* boxes    = (const float*)d_in[2];
  const int*   labels   = (const int*)d_in[3];
  const float* priors   = (const float*)d_in[4];

  char* ws = (char*)d_ws;
  const size_t NP = (size_t)NIMG * NPRI;
  int*   prior_obj = (int*)(ws);
  float* prior_iou = (float*)(ws + NP * 4);
  int*   prior_cls = (int*)(ws + NP * 8);
  float* neg_ce    = (float*)(ws + NP * 12);
  unsigned long long* objbest = (unsigned long long*)(ws + NP * 16);
  float* acc  = (float*)(ws + NP * 16 + (size_t)NIMG * NOBJ * 8);
  int*   npos = (int*)(ws + NP * 16 + (size_t)NIMG * NOBJ * 8 + 64);

  k_init<<<1, 256, 0, stream>>>(acc, npos, objbest);
  dim3 gm((NPRI + 255) / 256, NIMG);
  k_match<<<gm, 256, 0, stream>>>(boxes, priors, prior_obj, prior_iou, objbest);
  k_override<<<1, 32, 0, stream>>>(objbest, prior_obj, prior_iou);
  k_loc<<<gm, 256, 0, stream>>>(pred_off, boxes, labels, priors,
                                prior_obj, prior_iou, prior_cls, acc, npos);
  int ceBlocks = (int)((NP + CE_ROWS - 1) / CE_ROWS);
  k_ce<<<ceBlocks, 256, 0, stream>>>(scores, prior_cls, neg_ce, acc);
  k_topk<<<NIMG, 1024, 0, stream>>>(neg_ce, npos, acc);
  k_final<<<1, 1, 0, stream>>>(acc, npos, (float*)d_out);
}
